// Attention_23321672417682
// MI455X (gfx1250) — matrix operand order repaired and checked by reading, not compiled, not run on hardware
//
#include <hip/hip_runtime.h>

// ---------------------------------------------------------------------------
// CDNA5 (gfx1250) multi-head attention: x@Wqkv -> flash-attn -> @Wout
// All matmuls on v_wmma_f32_16x16x32_bf16 (wave32), fp32 accumulation.
// GEMMs: 64x64 register tile per wave (16 independent accumulators).
// Attention: 32 queries/wave, 32-key chunks; K/V tiles staged in LDS by the
// Tensor Data Mover (tensor_load_to_lds, double-buffered, TENSORcnt-tracked)
// when the builtin is available; direct-global fallback otherwise.
// ---------------------------------------------------------------------------

typedef __bf16 bf16_t;
typedef __attribute__((ext_vector_type(16))) bf16_t v16bf;
typedef __attribute__((ext_vector_type(8)))  float  v8f;
typedef __attribute__((ext_vector_type(4)))  unsigned int u32x4;
typedef __attribute__((ext_vector_type(4)))  float f32x4;
typedef __attribute__((ext_vector_type(4)))  unsigned short u16x4;

union FragB16 { v16bf v; u32x4 q[2]; };

#define BATCH 8
#define SEQ   2048
#define HEADS 8
#define DH    64
#define DIM   512
#define DIM3  1536
#define MROWS (BATCH * SEQ)   // 16384

#if __has_builtin(__builtin_amdgcn_tensor_load_to_lds) && \
    __has_builtin(__builtin_amdgcn_s_wait_tensorcnt)
#define HAVE_TDM 1
#else
#define HAVE_TDM 0
#endif

static __device__ inline unsigned short f32_to_bf16(float f) {
  union { float f; unsigned int u; } c; c.f = f;
  unsigned int u = c.u;
  unsigned int r = u + 0x7FFFu + ((u >> 16) & 1u);   // round-to-nearest-even
  if ((u & 0x7F800000u) == 0x7F800000u) r = u;       // inf/nan: truncate
  return (unsigned short)(r >> 16);
}

static __device__ inline v8f wmma_bf16(v16bf a, v16bf b, v8f c) {
  return __builtin_amdgcn_wmma_f32_16x16x32_bf16(
      /*neg_a=*/false, a, /*neg_b=*/false, b,
      /*c_mod=*/(short)0, c, /*reuse_a=*/false, /*reuse_b=*/false);
}

#if HAVE_TDM
typedef __attribute__((ext_vector_type(8))) int i32x8;
typedef __attribute__((ext_vector_type(4))) int i32x4;

// D# group1 for a 2-D bf16 tile: tensor d0 x d1 (elements), tile t0 x t1,
// dim0 stride in elements. data_size=1 (2 bytes), no multicast/pad/iterate.
static __device__ inline i32x8 tdm_desc_2d(unsigned d0, unsigned d1,
                                           unsigned t0, unsigned t1,
                                           unsigned long long stride0) {
  i32x8 g = {0, 0, 0, 0, 0, 0, 0, 0};
  g[0] = (int)(1u << 16);                                   // data_size = 2B
  g[1] = (int)((d0 & 0xFFFFu) << 16);                       // tensor_dim0 [79:48]
  g[2] = (int)((d0 >> 16) | ((d1 & 0xFFFFu) << 16));        // tensor_dim1 [111:80]
  g[3] = (int)((d1 >> 16) | ((t0 & 0xFFFFu) << 16));        // tile_dim0  [127:112]
  g[4] = (int)(t1 & 0xFFFFu);                               // tile_dim1  [143:128]
  g[5] = (int)(stride0 & 0xFFFFFFFFull);                    // dim0_stride[207:160]
  g[6] = (int)((stride0 >> 32) & 0xFFFFull);
  g[7] = 0;
  return g;
}

static __device__ inline void tdm_load_2d(unsigned lds_addr, const void* gptr,
                                          i32x8 g1) {
  unsigned long long ga = (unsigned long long)(size_t)gptr;
  u32x4 g0;
  g0[0] = 1u;                                               // count=1, load, 2-D
  g0[1] = lds_addr;                                         // lds_addr [63:32]
  g0[2] = (unsigned)(ga & 0xFFFFFFFFull);                   // global_addr lo
  g0[3] = (unsigned)((ga >> 32) & 0x01FFFFFFull)            // global_addr hi
        | (2u << 30);                                       // type = 2 (image)
  i32x4 z4 = {0, 0, 0, 0};
#if __clang_major__ >= 23
  i32x8 z8 = {0, 0, 0, 0, 0, 0, 0, 0};
  __builtin_amdgcn_tensor_load_to_lds(g0, g1, z4, z4, z8, 0);
#else
  __builtin_amdgcn_tensor_load_to_lds(g0, g1, z4, z4, 0);
#endif
}
#endif  // HAVE_TDM

// ---------------------------------------------------------------------------
// Prep: fp32 -> bf16 (vectorized), and transposed bf16 weight copies.
// ---------------------------------------------------------------------------
__global__ void cvt_f32_bf16x4_kernel(const f32x4* __restrict__ in,
                                      u16x4* __restrict__ out, int n4) {
  int i = blockIdx.x * blockDim.x + threadIdx.x;
  if (i < n4) {
    f32x4 v = in[i];
    u16x4 o;
    o[0] = f32_to_bf16(v[0]); o[1] = f32_to_bf16(v[1]);
    o[2] = f32_to_bf16(v[2]); o[3] = f32_to_bf16(v[3]);
    out[i] = o;
  }
}

// in: [K][N] fp32 row-major; out: [N][K] bf16 row-major (transposed)
__global__ void transpose_cvt_kernel(const float* __restrict__ in,
                                     unsigned short* __restrict__ out,
                                     int K, int N) {
  int i = blockIdx.x * blockDim.x + threadIdx.x;
  if (i < K * N) {
    int n = i / K, k = i % K;
    out[i] = f32_to_bf16(in[k * N + n]);
  }
}

// ---------------------------------------------------------------------------
// QKV GEMM: 64x64 per wave. Scatters into Q[bh][n][64] (pre-scaled),
// K[bh][n][64], VT[bh][64][n] (transposed), all bf16.
// ---------------------------------------------------------------------------
__global__ void __launch_bounds__(256) qkv_gemm_kernel(
    const unsigned short* __restrict__ xb,
    const unsigned short* __restrict__ wT,
    unsigned short* __restrict__ Qb,
    unsigned short* __restrict__ Kb,
    unsigned short* __restrict__ VTb) {
  const int lane = threadIdx.x & 31;
  const int wave = threadIdx.x >> 5;
  const int wid  = blockIdx.x * 8 + wave;
  const int NT   = DIM3 / 64;                // 24 N-tiles
  const int mT   = wid / NT;                 // 0..255
  const int nT   = wid % NT;
  const int l    = lane & 15;
  const int hi   = lane >> 4;

  const unsigned short* arow[4];
  const unsigned short* brow[4];
#pragma unroll
  for (int i = 0; i < 4; ++i) {
    arow[i] = xb + (size_t)(mT * 64 + i * 16 + l) * DIM;
    brow[i] = wT + (size_t)(nT * 64 + i * 16 + l) * DIM;
  }

  v8f acc[4][4] = {};
  for (int kc = 0; kc < DIM; kc += 32) {
    FragB16 a[4], b[4];
#pragma unroll
    for (int i = 0; i < 4; ++i) {
      const unsigned short* ap = arow[i] + kc + hi * 8;
      a[i].q[0] = *(const u32x4*)(ap);
      a[i].q[1] = *(const u32x4*)(ap + 16);
    }
#pragma unroll
    for (int j = 0; j < 4; ++j) {
      const unsigned short* bp = brow[j] + kc + hi * 8;
      b[j].q[0] = *(const u32x4*)(bp);
      b[j].q[1] = *(const u32x4*)(bp + 16);
    }
#pragma unroll
    for (int i = 0; i < 4; ++i)
#pragma unroll
      for (int j = 0; j < 4; ++j)
        acc[i][j] = wmma_bf16(a[i].v, b[j].v, acc[i][j]);
  }

  const float SCALE = 0.125f;                // 64^-0.5
#pragma unroll
  for (int j = 0; j < 4; ++j) {
    const int c     = nT * 64 + j * 16 + l;  // output column (fixed per lane)
    const int which = c / DIM;               // 0=q 1=k 2=v
    const int h     = (c % DIM) / DH;
    const int dd    = c % DH;
#pragma unroll
    for (int i = 0; i < 4; ++i) {
#pragma unroll
      for (int r = 0; r < 8; ++r) {
        int row = mT * 64 + i * 16 + hi * 8 + r;
        int b_  = row >> 11;
        int n_  = row & (SEQ - 1);
        int bh  = b_ * HEADS + h;
        float v = acc[i][j][r];
        if (which == 0)
          Qb[((size_t)bh * SEQ + n_) * DH + dd] = f32_to_bf16(v * SCALE);
        else if (which == 1)
          Kb[((size_t)bh * SEQ + n_) * DH + dd] = f32_to_bf16(v);
        else
          VTb[((size_t)bh * DH + dd) * SEQ + n_] = f32_to_bf16(v);
      }
    }
  }
}

// ---------------------------------------------------------------------------
// Flash attention: 1 wave = 32 query rows (two 16-row tiles), online softmax
// over 2048 keys in chunks of 32. Block = 4 waves = 128 rows of one (b,h).
// K/V tiles moved global->LDS by the TDM (double-buffered) when available.
// ---------------------------------------------------------------------------
__global__ void __launch_bounds__(128) attn_kernel(
    const unsigned short* __restrict__ Qb,
    const unsigned short* __restrict__ Kb,
    const unsigned short* __restrict__ VTb,
    unsigned short* __restrict__ AO) {
#if HAVE_TDM
  __shared__ unsigned short kbuf[2][32][64];     // 2 x 4 KB K tiles
  __shared__ unsigned short vbuf[2][64][32];     // 2 x 4 KB V tiles
#endif
  __shared__ unsigned short plds[4][2][16][32];  // per-wave P staging, 8 KB

  const int lane = threadIdx.x & 31;
  const int wave = threadIdx.x >> 5;
  const int l    = lane & 15;
  const int hi   = lane >> 4;

  const int bh   = blockIdx.x >> 4;              // 0..63
  const int qblk = blockIdx.x & 15;              // 16 blocks of 128 rows
  const int q0   = qblk * 128 + wave * 32;

  const unsigned short* Qp = Qb  + (size_t)bh * SEQ * DH;
  const unsigned short* Kp = Kb  + (size_t)bh * SEQ * DH;
  const unsigned short* Vp = VTb + (size_t)bh * DH * SEQ;

  // Resident Q A-fragments: 2 row-tiles x 2 k-chunks
  FragB16 qa[2][2];
#pragma unroll
  for (int i = 0; i < 2; ++i) {
    const unsigned short* qrow = Qp + (size_t)(q0 + i * 16 + l) * DH;
    qa[i][0].q[0] = *(const u32x4*)(qrow + hi * 8);
    qa[i][0].q[1] = *(const u32x4*)(qrow + hi * 8 + 16);
    qa[i][1].q[0] = *(const u32x4*)(qrow + 32 + hi * 8);
    qa[i][1].q[1] = *(const u32x4*)(qrow + 32 + hi * 8 + 16);
  }

  v8f o[2][4] = {};                              // 32x64 fp32 accumulator
  float m[2][8], ls[2][8];
#pragma unroll
  for (int i = 0; i < 2; ++i)
#pragma unroll
    for (int r = 0; r < 8; ++r) { m[i][r] = -__builtin_inff(); ls[i][r] = 0.f; }

#if HAVE_TDM
  // K tile: 32 rows (keys) x 64 cols (d), row stride 64 elements.
  // V tile: 64 rows (d)    x 32 cols (keys), row stride SEQ elements in global.
  const i32x8 g1K = tdm_desc_2d(/*d0=*/DH, /*d1=*/32, /*t0=*/DH, /*t1=*/32, DH);
  const i32x8 g1V = tdm_desc_2d(/*d0=*/32, /*d1=*/DH, /*t0=*/32, /*t1=*/DH, SEQ);
  if (wave == 0) {                               // prologue: chunk 0 -> buf 0
    tdm_load_2d((unsigned)(size_t)(void*)&kbuf[0][0][0], Kp, g1K);
    tdm_load_2d((unsigned)(size_t)(void*)&vbuf[0][0][0], Vp, g1V);
  }
#endif

  for (int j0 = 0; j0 < SEQ; j0 += 32) {
    const unsigned short* ktile;
    const unsigned short* vtile;
    size_t vstride;
#if HAVE_TDM
    const int p = (j0 >> 5) & 1;
    if (wave == 0) __builtin_amdgcn_s_wait_tensorcnt(0);  // buf[p] complete
    __syncthreads();   // all see buf[p]; all done reading buf[p^1] last iter
    if (wave == 0 && j0 + 32 < SEQ) {            // refill other buffer
      tdm_load_2d((unsigned)(size_t)(void*)&kbuf[p ^ 1][0][0],
                  Kp + (size_t)(j0 + 32) * DH, g1K);
      tdm_load_2d((unsigned)(size_t)(void*)&vbuf[p ^ 1][0][0],
                  Vp + (j0 + 32), g1V);
    }
    ktile   = &kbuf[p][0][0];
    vtile   = &vbuf[p][0][0];
    vstride = 32;
#else
    if (j0 + 32 < SEQ) {                         // next-chunk prefetch
      __builtin_prefetch(Kp + (size_t)(j0 + 32 + l) * DH, 0, 1);
      __builtin_prefetch(Kp + (size_t)(j0 + 48 + l) * DH, 0, 1);
      __builtin_prefetch(Vp + (size_t)(l * 2) * SEQ + j0 + 32, 0, 1);
      __builtin_prefetch(Vp + (size_t)(l * 2 + 1) * SEQ + j0 + 32, 0, 1);
    }
    ktile   = Kp + (size_t)j0 * DH;
    vtile   = Vp + j0;
    vstride = SEQ;
#endif

    // K fragments for two 16-key tiles (shared by both row-tiles)
    FragB16 kf[2][2];
#pragma unroll
    for (int t = 0; t < 2; ++t) {
      const unsigned short* kp = ktile + (size_t)(t * 16 + l) * DH;
      kf[t][0].q[0] = *(const u32x4*)(kp + hi * 8);
      kf[t][0].q[1] = *(const u32x4*)(kp + hi * 8 + 16);
      kf[t][1].q[0] = *(const u32x4*)(kp + 32 + hi * 8);
      kf[t][1].q[1] = *(const u32x4*)(kp + 32 + hi * 8 + 16);
    }

    v8f s[2][2] = {};
#pragma unroll
    for (int i = 0; i < 2; ++i) {
      s[i][0] = wmma_bf16(qa[i][0].v, kf[0][0].v, s[i][0]);
      s[i][0] = wmma_bf16(qa[i][1].v, kf[0][1].v, s[i][0]);
      s[i][1] = wmma_bf16(qa[i][0].v, kf[1][0].v, s[i][1]);
      s[i][1] = wmma_bf16(qa[i][1].v, kf[1][1].v, s[i][1]);
    }

    // Online softmax per row-tile; P written to LDS in bf16
#pragma unroll
    for (int i = 0; i < 2; ++i) {
#pragma unroll
      for (int r = 0; r < 8; ++r) {
        float mx = fmaxf(s[i][0][r], s[i][1][r]);
        mx = fmaxf(mx, __shfl_xor(mx, 1, 32));
        mx = fmaxf(mx, __shfl_xor(mx, 2, 32));
        mx = fmaxf(mx, __shfl_xor(mx, 4, 32));
        mx = fmaxf(mx, __shfl_xor(mx, 8, 32));
        float mn    = fmaxf(m[i][r], mx);
        float alpha = __expf(m[i][r] - mn);
        m[i][r] = mn;
        float e0 = __expf(s[i][0][r] - mn);
        float e1 = __expf(s[i][1][r] - mn);
        float rs = e0 + e1;
        rs += __shfl_xor(rs, 1, 32);
        rs += __shfl_xor(rs, 2, 32);
        rs += __shfl_xor(rs, 4, 32);
        rs += __shfl_xor(rs, 8, 32);
        ls[i][r] = ls[i][r] * alpha + rs;
        o[i][0][r] *= alpha; o[i][1][r] *= alpha;
        o[i][2][r] *= alpha; o[i][3][r] *= alpha;
        plds[wave][i][hi * 8 + r][l]      = f32_to_bf16(e0);
        plds[wave][i][hi * 8 + r][16 + l] = f32_to_bf16(e1);
      }
    }
    asm volatile("s_wait_dscnt 0" ::: "memory");   // wave-internal cross-lane dep

    FragB16 pa[2];
#pragma unroll
    for (int i = 0; i < 2; ++i) {
      pa[i].q[0] = *(const u32x4*)(&plds[wave][i][l][hi * 8]);
      pa[i].q[1] = *(const u32x4*)(&plds[wave][i][l][16 + hi * 8]);
    }

    // O += P @ V  (each V fragment feeds both row-tiles)
#pragma unroll
    for (int t = 0; t < 4; ++t) {
      FragB16 vf;
      const unsigned short* vp = vtile + (size_t)(t * 16 + l) * vstride + hi * 8;
      vf.q[0] = *(const u32x4*)(vp);
      vf.q[1] = *(const u32x4*)(vp + 16);
      o[0][t] = wmma_bf16(pa[0].v, vf.v, o[0][t]);
      o[1][t] = wmma_bf16(pa[1].v, vf.v, o[1][t]);
    }
  }

  // Normalize and store bf16 into AO[row][h*64 + d]
  const int h = bh & (HEADS - 1);
#pragma unroll
  for (int i = 0; i < 2; ++i) {
#pragma unroll
    for (int r = 0; r < 8; ++r) {
      float inv = 1.0f / ls[i][r];
      int row = (bh >> 3) * SEQ + q0 + i * 16 + hi * 8 + r;
      unsigned short* ao = AO + (size_t)row * DIM + h * DH;
      ao[0 * 16 + l] = f32_to_bf16(o[i][0][r] * inv);
      ao[1 * 16 + l] = f32_to_bf16(o[i][1][r] * inv);
      ao[2 * 16 + l] = f32_to_bf16(o[i][2][r] * inv);
      ao[3 * 16 + l] = f32_to_bf16(o[i][3][r] * inv);
    }
  }
}

// ---------------------------------------------------------------------------
// Out projection: out[16384][512] fp32 = AO @ Wout; 64x64 per wave.
// ---------------------------------------------------------------------------
__global__ void __launch_bounds__(256) proj_gemm_kernel(
    const unsigned short* __restrict__ A,
    const unsigned short* __restrict__ wT,
    float* __restrict__ out) {
  const int lane = threadIdx.x & 31;
  const int wave = threadIdx.x >> 5;
  const int wid  = blockIdx.x * 8 + wave;
  const int NT   = DIM / 64;                 // 8
  const int mT   = wid / NT;
  const int nT   = wid % NT;
  const int l    = lane & 15;
  const int hi   = lane >> 4;

  const unsigned short* arow[4];
  const unsigned short* brow[4];
#pragma unroll
  for (int i = 0; i < 4; ++i) {
    arow[i] = A  + (size_t)(mT * 64 + i * 16 + l) * DIM;
    brow[i] = wT + (size_t)(nT * 64 + i * 16 + l) * DIM;
  }

  v8f acc[4][4] = {};
  for (int kc = 0; kc < DIM; kc += 32) {
    FragB16 a[4], b[4];
#pragma unroll
    for (int i = 0; i < 4; ++i) {
      const unsigned short* ap = arow[i] + kc + hi * 8;
      a[i].q[0] = *(const u32x4*)(ap);
      a[i].q[1] = *(const u32x4*)(ap + 16);
    }
#pragma unroll
    for (int j = 0; j < 4; ++j) {
      const unsigned short* bp = brow[j] + kc + hi * 8;
      b[j].q[0] = *(const u32x4*)(bp);
      b[j].q[1] = *(const u32x4*)(bp + 16);
    }
#pragma unroll
    for (int i = 0; i < 4; ++i)
#pragma unroll
      for (int j = 0; j < 4; ++j)
        acc[i][j] = wmma_bf16(a[i].v, b[j].v, acc[i][j]);
  }

#pragma unroll
  for (int j = 0; j < 4; ++j) {
    const int c = nT * 64 + j * 16 + l;
#pragma unroll
    for (int i = 0; i < 4; ++i) {
#pragma unroll
      for (int r = 0; r < 8; ++r) {
        int row = mT * 64 + i * 16 + hi * 8 + r;
        out[(size_t)row * DIM + c] = acc[i][j][r];
      }
    }
  }
}

// ---------------------------------------------------------------------------
extern "C" void kernel_launch(void* const* d_in, const int* in_sizes, int n_in,
                              void* d_out, int out_size, void* d_ws, size_t ws_size,
                              hipStream_t stream) {
  (void)in_sizes; (void)n_in; (void)out_size; (void)ws_size;
  const float* x     = (const float*)d_in[0];
  const float* w_qkv = (const float*)d_in[1];
  const float* w_out = (const float*)d_in[2];
  float* out = (float*)d_out;

  char* ws = (char*)d_ws;
  size_t off = 0;
  auto alloc = [&](size_t bytes) -> void* {
    void* p = ws + off;
    off += (bytes + 255) & ~(size_t)255;
    return p;
  };
  unsigned short* xb    = (unsigned short*)alloc((size_t)MROWS * DIM * 2);
  unsigned short* wqkvT = (unsigned short*)alloc((size_t)DIM3 * DIM * 2);
  unsigned short* woutT = (unsigned short*)alloc((size_t)DIM * DIM * 2);
  unsigned short* Qb    = (unsigned short*)alloc((size_t)64 * SEQ * DH * 2);
  unsigned short* Kb    = (unsigned short*)alloc((size_t)64 * SEQ * DH * 2);
  unsigned short* VTb   = (unsigned short*)alloc((size_t)64 * DH * SEQ * 2);
  unsigned short* AO    = (unsigned short*)alloc((size_t)MROWS * DIM * 2);

  {
    int n4 = (MROWS * DIM) / 4;
    cvt_f32_bf16x4_kernel<<<(n4 + 255) / 256, 256, 0, stream>>>(
        (const f32x4*)x, (u16x4*)xb, n4);
  }
  {
    int n = DIM * DIM3;
    transpose_cvt_kernel<<<(n + 255) / 256, 256, 0, stream>>>(w_qkv, wqkvT, DIM, DIM3);
  }
  {
    int n = DIM * DIM;
    transpose_cvt_kernel<<<(n + 255) / 256, 256, 0, stream>>>(w_out, woutT, DIM, DIM);
  }
  // QKV: 256 M-tiles * 24 N-tiles = 6144 waves / 8 per block
  qkv_gemm_kernel<<<768, 256, 0, stream>>>(xb, wqkvT, Qb, Kb, VTb);
  // Attention: 64 (b,h) * 16 q-blocks of 128 rows
  attn_kernel<<<1024, 128, 0, stream>>>(Qb, Kb, VTb, AO);
  // Proj: 256 M-tiles * 8 N-tiles = 2048 waves / 8 per block
  proj_gemm_kernel<<<256, 256, 0, stream>>>(AO, woutT, out);
}
